// Attention_75505525063766
// MI455X (gfx1250) — matrix operand order repaired and checked by reading, not compiled, not run on hardware
//
#include <hip/hip_runtime.h>
#include <hip/hip_bf16.h>

typedef __attribute__((ext_vector_type(16))) __bf16 v16bf;
typedef __attribute__((ext_vector_type(8)))  __bf16 v8bf;
typedef __attribute__((ext_vector_type(4)))  __bf16 v4bf;
typedef __attribute__((ext_vector_type(8)))  float  v8f;
typedef __attribute__((ext_vector_type(4)))  float  v4f;
typedef __attribute__((ext_vector_type(2)))  float  v2f;

constexpr int kB = 4, kN = 1024, kD = 64, kH = 8, kDH = 64, kInner = 512, kC3 = 1536;

// ---------------- wave reductions (wave32) ----------------
__device__ inline float wred_sum32(float v) {
#pragma unroll
  for (int m = 16; m >= 1; m >>= 1) v += __shfl_xor(v, m, 32);
  return v;
}
__device__ inline float hred_max16(float v) {
#pragma unroll
  for (int m = 8; m >= 1; m >>= 1) v = fmaxf(v, __shfl_xor(v, m, 32));
  return v;
}
__device__ inline float hred_sum16(float v) {
#pragma unroll
  for (int m = 8; m >= 1; m >>= 1) v += __shfl_xor(v, m, 32);
  return v;
}
__device__ inline void wave_atomic_add(float* dst, float v) {
  v = wred_sum32(v);
  if ((threadIdx.x & 31) == 0) atomicAdd(dst, v);
}

__device__ inline v8f wmma_bf16(v16bf a, v16bf b, v8f c) {
  return __builtin_amdgcn_wmma_f32_16x16x32_bf16(false, a, false, b, (short)0, c, false, false);
}

// 16-byte vector loads for WMMA fragments (works for global and LDS pointers)
__device__ inline v16bf ld2x8(const __bf16* p0, const __bf16* p1) {
  v8bf lo = *(const v8bf*)p0;
  v8bf hi = *(const v8bf*)p1;
  v16bf r;
#pragma unroll
  for (int i = 0; i < 8; ++i) { r[i] = lo[i]; r[i + 8] = hi[i]; }
  return r;
}

__device__ inline float elu1(float v) { return v > 0.f ? v + 1.f : __expf(v); }

// ---------------- init ----------------
__global__ void init_kernel(float* scal) {
  if (threadIdx.x < 16) scal[threadIdx.x] = 0.f;
}

// ---------------- per-row squared norms of x ----------------
__global__ void sq_kernel(const float* x, float* sq) {
  int idx = blockIdx.x * blockDim.x + threadIdx.x;  // B*N
  if (idx < kB * kN) {
    const float* r = x + (size_t)idx * kD;
    float s = 0.f;
#pragma unroll
    for (int d = 0; d < kD; ++d) { float t = r[d]; s += t * t; }
    sq[idx] = s;
  }
}

// ---------------- weight pre-transpose to bf16 (k-contiguous B fragments) ----------------
__global__ void wprep_kernel(const float* Wqkv, const float* Wout,
                             __bf16* wqkvT, __bf16* woutT) {
  int g = blockIdx.x * 256 + threadIdx.x;
  if (g < kD * kC3) {                 // Wqkv: (64 x 1536) -> wqkvT[c][k]
    int k = g / kC3, c = g % kC3;
    wqkvT[(size_t)c * kD + k] = (__bf16)Wqkv[g];
  }
  if (g < kInner * kD) {              // Wout: (512 x 64) -> woutT[c][k]
    int k = g / kD, c = g % kD;
    woutT[(size_t)c * kInner + k] = (__bf16)Wout[g];
  }
}

// ---------------- QKV projection: bf16 WMMA, fused bias+elu+head reorder ----------------
// qe,ke stored (b,h,n,d) row-major; v stored TRANSPOSED (b,h,d,n) for PV B-fragments.
__global__ __launch_bounds__(32) void qkv_kernel(const float* x, const __bf16* wqkvT,
                                                 const float* bias,
                                                 __bf16* qe, __bf16* ke, __bf16* vv) {
  int lane = threadIdx.x;
  int r0 = blockIdx.x << 4, c0 = blockIdx.y << 6;
  int n15 = lane & 15; int hi = (lane >= 16);
  const float* ar = x + (size_t)(r0 + n15) * kD;
  v16bf a0, a1;
  {
    int off0 = hi ? 8 : 0;
#pragma unroll
    for (int i = 0; i < 8; ++i) {
      a0[i] = (__bf16)ar[off0 + i];       a0[i + 8] = (__bf16)ar[off0 + 16 + i];
      a1[i] = (__bf16)ar[32 + off0 + i];  a1[i + 8] = (__bf16)ar[32 + off0 + 16 + i];
    }
  }
  int koffB = hi ? 8 : 0;
  // wave-uniform row decomposition (16-row tile never straddles a batch)
  int bidx = r0 >> 10;
  int nbase = (r0 & 1023) + (hi ? 8 : 0);
  for (int cc = 0; cc < 4; ++cc) {
    int c = c0 + (cc << 4);
    const __bf16* wrow = wqkvT + (size_t)(c + n15) * kD;
    v16bf b0 = ld2x8(wrow + koffB, wrow + koffB + 16);
    v16bf b1 = ld2x8(wrow + 32 + koffB, wrow + 32 + koffB + 16);
    v8f acc = {};
    acc = wmma_bf16(a0, b0, acc);
    acc = wmma_bf16(a1, b1, acc);
    // wave-uniform: section & head (chunk of 16 cols never straddles them)
    int sec = c >> 9;                 // 0=q 1=k 2=v
    int h = (c & 511) >> 6;
    int d = (c & 63) + n15;
    float bv = bias[c + n15];
    if (sec == 2) {                   // v transposed: (b,h,d,n); 8 consecutive n -> one b128 store
      v8bf pv;
#pragma unroll
      for (int r = 0; r < 8; ++r) pv[r] = (__bf16)(acc[r] + bv);
      *(v8bf*)(vv + ((size_t)((bidx * kH + h) * kDH + d)) * kN + nbase) = pv;
    } else {
      __bf16* dst = (sec == 0 ? qe : ke);
      size_t base = ((size_t)((bidx * kH + h) * kN + nbase)) * kDH + d;
#pragma unroll
      for (int r = 0; r < 8; ++r) dst[base + (size_t)r * kDH] = (__bf16)elu1(acc[r] + bv);
    }
  }
}

// ---------------- spe: d2 = |xi|^2+|xj|^2-2 xi.xj via fp32 WMMA; sum(dist) ----------------
__global__ __launch_bounds__(32) void spe_d2_kernel(const float* x, const float* sq,
                                                    float* spe, float* scal) {
  int lane = threadIdx.x;
  int j0 = blockIdx.x << 4, i0 = blockIdx.y << 4, b = blockIdx.z;
  int n15 = lane & 15; int hi = (lane >= 16);
  const float* xb = x + (size_t)b * kN * kD;
  const float* ar = xb + (size_t)(i0 + n15) * kD;
  const float* br = xb + (size_t)(j0 + n15) * kD;
  int ko = hi ? 2 : 0;
  v8f acc = {};
#pragma unroll
  for (int k = 0; k < kD; k += 4) {
    v2f a = { ar[k + ko], ar[k + ko + 1] };
    v2f bb = { br[k + ko], br[k + ko + 1] };
    acc = __builtin_amdgcn_wmma_f32_16x16x4_f32(false, a, false, bb, (short)0, acc, false, false);
  }
  const float* sqb = sq + (size_t)b * kN;
  float* speB = spe + (size_t)b * kN * kN;
  int j = j0 + n15;
  float sqj = sqb[j];
  float dsum = 0.f;
#pragma unroll
  for (int r = 0; r < 8; ++r) {
    int i = i0 + (hi ? 8 : 0) + r;
    float d2 = sqb[i] + sqj - 2.f * acc[r];
    d2 = fmaxf(d2, 0.f);
    speB[(size_t)i * kN + j] = d2;
    dsum += (d2 > 1e-12f) ? sqrtf(d2) : 0.f;
  }
  dsum = wred_sum32(dsum);
  if (lane == 0) atomicAdd(&scal[0], dsum);
}

// ---------------- spa grid distances ----------------
__device__ inline float spa_d2(int i, int j) {
  int ix = i >> 6, iy = (i >> 2) & 15, iz = i & 3;
  int jx = j >> 6, jy = (j >> 2) & 15, jz = j & 3;
  float dx = (float)(ix - jx), dy = (float)(iy - jy), dz = (float)(iz - jz);
  return dx * dx + dy * dy + dz * dz;
}
__global__ void spa_sum_kernel(float* scal) {
  int g = blockIdx.x * 256 + threadIdx.x;  // N*N exactly
  float d2 = spa_d2(g >> 10, g & 1023);
  float dist = (d2 > 1e-12f) ? sqrtf(d2) : 0.f;
  wave_atomic_add(&scal[1], dist);
}
__global__ void fin1_kernel(float* scal) {
  if (threadIdx.x == 0) {
    float s_spe = scal[0] / (float)((size_t)kB * kN * kN);
    float s_spa = scal[1] / (float)((size_t)kN * kN);
    scal[2] = 1.f / (2.f * s_spe * s_spe);
    scal[3] = 1.f / (2.f * s_spa * s_spa);
  }
}
__global__ void spe_w_kernel(float* spe, float* scal) {
  size_t g = (size_t)blockIdx.x * 256 + threadIdx.x;  // B*N*N exactly
  float inv = scal[2];
  float d2 = spe[g];
  float w = __expf(-d2 * inv);
  spe[g] = w;
  wave_atomic_add(&scal[4], w);
}
__global__ void spa_w_kernel(float* spa, float* scal) {
  int g = blockIdx.x * 256 + threadIdx.x;  // N*N exactly
  float inv = scal[3];
  float d2 = spa_d2(g >> 10, g & 1023);
  float w = __expf(-d2 * inv);
  spa[g] = w;
  wave_atomic_add(&scal[5], w);
}
__global__ void fin2_kernel(float* scal) {
  if (threadIdx.x == 0) {
    scal[6] = scal[4] / (float)((size_t)kB * kN * kN);
    scal[7] = scal[5] / (float)((size_t)kN * kN);
  }
}

// masked score helper: loads 8 consecutive mask values using matrix symmetry
__device__ inline void load_masks8(const float* speB, const float* spa,
                                   int j, int irowbase, float spe_mean, float spa_mean,
                                   float* sv, float* av) {
  const float* sprow = speB + (size_t)j * kN + irowbase;   // spe symmetric: [j][i]
  const float* aprow = spa + (size_t)j * kN + irowbase;    // spa symmetric
  v4f s0 = *(const v4f*)sprow, s1 = *(const v4f*)(sprow + 4);
  v4f a0 = *(const v4f*)aprow, a1 = *(const v4f*)(aprow + 4);
#pragma unroll
  for (int r = 0; r < 4; ++r) {
    sv[r] = (s0[r] < spe_mean) ? 0.f : s0[r];
    sv[r + 4] = (s1[r] < spe_mean) ? 0.f : s1[r];
    av[r] = (a0[r] < spa_mean) ? 0.f : a0[r];
    av[r + 4] = (a1[r] < spa_mean) ? 0.f : a1[r];
  }
}

// ---------------- flash-style attention: two-pass softmax + PV, bf16 WMMA ----------------
__global__ __launch_bounds__(32) void attn_kernel(const __bf16* qe, const __bf16* ke, const __bf16* vv,
                                                  const float* spe, const float* spa,
                                                  const float* scal, float* ao) {
  const int lane = threadIdx.x;
  const int bh = blockIdx.x;        // 0..31
  const int i0 = blockIdx.y << 4;   // row tile
  const int b = bh >> 3, h = bh & 7;
  const __bf16* qb = qe + (size_t)bh * kN * kDH;
  const __bf16* kb = ke + (size_t)bh * kN * kDH;
  const __bf16* vb = vv + (size_t)bh * kDH * kN;   // transposed (d,n)
  const float* speB = spe + (size_t)b * kN * kN;
  const float spe_mean = scal[6];
  const float spa_mean = scal[7];
  const int n15 = lane & 15; const int hi = (lane >= 16);
  const int koffB = hi ? 16 : 0;
  const int irowbase = i0 + (hi ? 8 : 0);
  const int off0 = hi ? 8 : 0;

  // A = qe rows (16x64)
  const __bf16* arow = qb + (size_t)(i0 + n15) * kDH;
  v16bf a0 = ld2x8(arow + koffB, arow + koffB + 8);
  v16bf a1 = ld2x8(arow + 32 + koffB, arow + 40 + koffB);

  float m[8], ssum[8];
#pragma unroll
  for (int r = 0; r < 8; ++r) { m[r] = -1e30f; ssum[r] = 0.f; }

  // Pass A: per-row max & exp-sum
  for (int j0 = 0; j0 < kN; j0 += 16) {
    const __bf16* brow = kb + (size_t)(j0 + n15) * kDH;
    int jpf = (j0 + 16 < kN) ? (j0 + 16) : j0;
    __builtin_prefetch(kb + (size_t)(jpf + n15) * kDH, 0, 1);
    v16bf b0 = ld2x8(brow + koffB, brow + koffB + 8);
    v16bf b1 = ld2x8(brow + 32 + koffB, brow + 32 + koffB + 8);
    v8f c = {};
    c = wmma_bf16(a0, b0, c);
    c = wmma_bf16(a1, b1, c);
    int j = j0 + n15;
    float sv[8], av[8];
    load_masks8(speB, spa, j, irowbase, spe_mean, spa_mean, sv, av);
#pragma unroll
    for (int r = 0; r < 8; ++r) {
      float val = c[r] * 0.125f * sv[r] * av[r];
      float tmax = hred_max16(val);
      float nm = fmaxf(m[r], tmax);
      float tsum = hred_sum16(__expf(val - nm));
      ssum[r] = ssum[r] * __expf(m[r] - nm) + tsum;
      m[r] = nm;
    }
  }
  float inv_s[8];
#pragma unroll
  for (int r = 0; r < 8; ++r) inv_s[r] = 1.f / ssum[r];

  // Pass B: recompute, normalize, transpose P via LDS, PV WMMA
  __shared__ __bf16 Pl[16][40];     // 80B rows -> 16B-aligned b128 fragments
  v8f z = {};
  v8f o0 = z, o1 = z, o2 = z, o3 = z;
  for (int j0 = 0; j0 < kN; j0 += 32) {
#pragma unroll
    for (int t = 0; t < 2; ++t) {
      int jt = j0 + t * 16;
      const __bf16* brow = kb + (size_t)(jt + n15) * kDH;
      v16bf b0 = ld2x8(brow + koffB, brow + koffB + 8);
      v16bf b1 = ld2x8(brow + 32 + koffB, brow + 32 + koffB + 8);
      v8f c = {};
      c = wmma_bf16(a0, b0, c);
      c = wmma_bf16(a1, b1, c);
      int j = jt + n15;
      float sv[8], av[8];
      load_masks8(speB, spa, j, irowbase, spe_mean, spa_mean, sv, av);
#pragma unroll
      for (int r = 0; r < 8; ++r) {
        float val = c[r] * 0.125f * sv[r] * av[r];
        float p = __expf(val - m[r]) * inv_s[r];
        Pl[(hi ? 8 : 0) + r][t * 16 + n15] = (__bf16)p;
      }
    }
    __syncthreads();
    v16bf pa = ld2x8(&Pl[n15][koffB], &Pl[n15][koffB + 8]);
    const __bf16* v0 = vb + (size_t)n15 * kN + j0 + koffB;           // d0 = 0
    const __bf16* v1 = vb + (size_t)(16 + n15) * kN + j0 + koffB;    // d0 = 16
    const __bf16* v2 = vb + (size_t)(32 + n15) * kN + j0 + koffB;    // d0 = 32
    const __bf16* v3 = vb + (size_t)(48 + n15) * kN + j0 + koffB;    // d0 = 48
    o0 = wmma_bf16(pa, ld2x8(v0, v0 + 8), o0);
    o1 = wmma_bf16(pa, ld2x8(v1, v1 + 8), o1);
    o2 = wmma_bf16(pa, ld2x8(v2, v2 + 8), o2);
    o3 = wmma_bf16(pa, ld2x8(v3, v3 + 8), o3);
    __syncthreads();
  }
#pragma unroll
  for (int r = 0; r < 8; ++r) {
    int row = (hi ? 8 : 0) + r;
    float* dst = ao + ((size_t)(b * kN + i0 + row)) * kInner + h * 64 + n15;
    dst[0]  = o0[r];
    dst[16] = o1[r];
    dst[32] = o2[r];
    dst[48] = o3[r];
  }
}

// ---------------- LayerNorm -> GELU(exact) -> output GEMM (bf16 WMMA) ----------------
__global__ __launch_bounds__(32) void out_kernel(const float* ao, const float* lng, const float* lnb,
                                                 const __bf16* woutT, const float* bout, float* y) {
  int lane = threadIdx.x;
  int r0 = blockIdx.x << 4;
  __shared__ __bf16 gb[16][512];
  for (int r = 0; r < 16; ++r) {
    const float* row = ao + (size_t)(r0 + r) * kInner;
    float s = 0.f, s2 = 0.f;
#pragma unroll
    for (int it = 0; it < 4; ++it) {
      v4f t = *(const v4f*)(row + lane * 4 + it * 128);
#pragma unroll
      for (int q = 0; q < 4; ++q) { s += t[q]; s2 += t[q] * t[q]; }
    }
    s = wred_sum32(s); s2 = wred_sum32(s2);
    float mu = s * (1.f / kInner);
    float var = s2 * (1.f / kInner) - mu * mu;
    float rinv = rsqrtf(var + 1e-5f);
#pragma unroll
    for (int it = 0; it < 4; ++it) {
      int c = lane * 4 + it * 128;
      v4f t = *(const v4f*)(row + c);
      v4f g = *(const v4f*)(lng + c);
      v4f bta = *(const v4f*)(lnb + c);
      v4bf outp;
#pragma unroll
      for (int q = 0; q < 4; ++q) {
        float u = g[q] * (t[q] - mu) * rinv + bta[q];
        outp[q] = (__bf16)(0.5f * u * (1.f + erff(u * 0.70710678118f)));
      }
      *(v4bf*)&gb[r][c] = outp;
    }
  }
  __syncthreads();
  int n15 = lane & 15; int hi = (lane >= 16);
  int koffB = hi ? 16 : 0;
  int off0 = hi ? 8 : 0;
  const __bf16* w0 = woutT + (size_t)n15 * kInner;         // col n15
  const __bf16* w1 = woutT + (size_t)(16 + n15) * kInner;
  const __bf16* w2 = woutT + (size_t)(32 + n15) * kInner;
  const __bf16* w3 = woutT + (size_t)(48 + n15) * kInner;
  v8f z = {}; v8f o0 = z, o1 = z, o2 = z, o3 = z;
  for (int k = 0; k < kInner; k += 32) {
    v16bf a = ld2x8(&gb[n15][k + koffB], &gb[n15][k + koffB + 8]);
    o0 = wmma_bf16(a, ld2x8(w0 + k + koffB, w0 + k + koffB + 8), o0);
    o1 = wmma_bf16(a, ld2x8(w1 + k + koffB, w1 + k + koffB + 8), o1);
    o2 = wmma_bf16(a, ld2x8(w2 + k + koffB, w2 + k + koffB + 8), o2);
    o3 = wmma_bf16(a, ld2x8(w3 + k + koffB, w3 + k + koffB + 8), o3);
  }
#pragma unroll
  for (int r = 0; r < 8; ++r) {
    int row = r0 + (hi ? 8 : 0) + r;
    float* dst = y + (size_t)row * kD;
    dst[n15]      = o0[r] + bout[n15];
    dst[16 + n15] = o1[r] + bout[16 + n15];
    dst[32 + n15] = o2[r] + bout[32 + n15];
    dst[48 + n15] = o3[r] + bout[48 + n15];
  }
}

// ---------------- host launcher ----------------
extern "C" void kernel_launch(void* const* d_in, const int* in_sizes, int n_in,
                              void* d_out, int out_size, void* d_ws, size_t ws_size,
                              hipStream_t stream) {
  (void)in_sizes; (void)n_in; (void)out_size; (void)ws_size;
  const float* x    = (const float*)d_in[0];
  const float* Wqkv = (const float*)d_in[1];
  const float* bqkv = (const float*)d_in[2];
  const float* lng  = (const float*)d_in[3];
  const float* lnb  = (const float*)d_in[4];
  const float* Wout = (const float*)d_in[5];
  const float* bout = (const float*)d_in[6];
  float* y = (float*)d_out;

  char* ws = (char*)d_ws;
  const size_t MB4 = 4194304;
  float*  scal  = (float*)(ws + 0);
  float*  sq    = (float*)(ws + 4096);
  __bf16* qe    = (__bf16*)(ws + 32768);
  __bf16* ke    = (__bf16*)(ws + 32768 + MB4);
  __bf16* vv    = (__bf16*)(ws + 32768 + 2 * MB4);
  float*  spe   = (float*)(ws + 32768 + 3 * MB4);
  float*  spa   = (float*)(ws + 32768 + 3 * MB4 + 16777216);
  float*  ao    = (float*)(ws + 32768 + 4 * MB4 + 16777216);
  __bf16* wqkvT = (__bf16*)(ws + 32768 + 6 * MB4 + 16777216);
  __bf16* woutT = (__bf16*)(ws + 32768 + 6 * MB4 + 16777216 + 262144);

  init_kernel<<<1, 32, 0, stream>>>(scal);
  sq_kernel<<<16, 256, 0, stream>>>(x, sq);
  wprep_kernel<<<384, 256, 0, stream>>>(Wqkv, Wout, wqkvT, woutT);
  qkv_kernel<<<dim3(256, 24), 32, 0, stream>>>(x, wqkvT, bqkv, qe, ke, vv);
  spe_d2_kernel<<<dim3(64, 64, 4), 32, 0, stream>>>(x, sq, spe, scal);
  spa_sum_kernel<<<4096, 256, 0, stream>>>(scal);
  fin1_kernel<<<1, 32, 0, stream>>>(scal);
  spe_w_kernel<<<16384, 256, 0, stream>>>(spe, scal);
  spa_w_kernel<<<4096, 256, 0, stream>>>(spa, scal);
  fin2_kernel<<<1, 32, 0, stream>>>(scal);
  attn_kernel<<<dim3(32, 64), 32, 0, stream>>>(qe, ke, vv, spe, spa, scal, ao);
  out_kernel<<<256, 32, 0, stream>>>(ao, lng, lnb, woutT, bout, y);
}
